// Net_42356967473780
// MI455X (gfx1250) — matrix operand order repaired and checked by reading, not compiled, not run on hardware
//
#include <hip/hip_runtime.h>
#include <hip/hip_bf16.h>
#include <math.h>

typedef __attribute__((ext_vector_type(16))) __bf16 v16bf;
typedef __attribute__((ext_vector_type(8)))  float  v8f;

#define NB    32          // batch
#define C4C   2048        // x4 channels
#define CFC   512         // hie_fea channels
#define KCLS  20          // foreground classes
#define HWP   1024        // H*W
#define WDIM  32
#define NSEED 5
#define NITER 10

// ---------------------------------------------------------------------------
// Kernel 1: cam[n,k,p] = sum_c x4[n,c,p] * w[k,c]   (WMMA bf16, f32 accum)
// grid (8, 32), block 256 (8 waves). wave -> one 16-pixel tile, 2 M-tiles.
// ---------------------------------------------------------------------------
__global__ __launch_bounds__(256) void cam_gemm_kernel(
    const float* __restrict__ x4, const float* __restrict__ clsw,
    float* __restrict__ cam)
{
  const int lane  = threadIdx.x & 31;
  const int wv    = threadIdx.x >> 5;
  const int n     = blockIdx.y;
  const int ptile = blockIdx.x * 8 + wv;     // 0..63
  const int m     = lane & 15;
  const int hi    = lane >> 4;               // half-wave selector
  const int pix   = ptile * 16 + m;

  v8f acc0 = {};  // rows 0..15
  v8f acc1 = {};  // rows 16..31 (only 16..19 valid)

  for (int c0 = 0; c0 < C4C; c0 += 32) {
    // B fragment (32x16 K x N): lane holds pixel column `pix`;
    // element j<8 <-> K = c0 + hi*8 + j, j>=8 <-> K = c0 + hi*8 + 16 + (j-8)   (f32 -> bf16 on the fly)
    v16bf b;
    const float* xp = x4 + ((size_t)n * C4C + c0 + hi * 8) * HWP + pix;
#pragma unroll
    for (int j = 0; j < 16; ++j) b[j] = (__bf16)xp[(size_t)(j < 8 ? j : j + 8) * HWP];

    // A fragments (16x32 M x K): lanes 0-15 row m, elems j<8 -> K=c0+hi*8+j,
    // elems j>=8 -> K=c0+hi*8+16+(j-8)  (per ISA 16-bit A layout table)
    const int kb = c0 + hi * 8;
    v16bf a0, a1;
    {
      const float* w0 = clsw + (size_t)m * C4C;       // row m (< 20 always)
#pragma unroll
      for (int j = 0; j < 8; ++j) {
        a0[j]     = (__bf16)w0[kb + j];
        a0[j + 8] = (__bf16)w0[kb + 16 + j];
      }
      const int k1 = 16 + m;
      if (k1 < KCLS) {
        const float* w1 = clsw + (size_t)k1 * C4C;
#pragma unroll
        for (int j = 0; j < 8; ++j) {
          a1[j]     = (__bf16)w1[kb + j];
          a1[j + 8] = (__bf16)w1[kb + 16 + j];
        }
      } else {
#pragma unroll
        for (int j = 0; j < 16; ++j) a1[j] = (__bf16)0.f;
      }
    }
    acc0 = __builtin_amdgcn_wmma_f32_16x16x32_bf16(false, a0, false, b,
                                                   (short)0, acc0, false, false);
    acc1 = __builtin_amdgcn_wmma_f32_16x16x32_bf16(false, a1, false, b,
                                                   (short)0, acc1, false, false);
  }

  // C layout: VGPR r -> row M = r + 8*hi, col N = lane&15
#pragma unroll
  for (int r = 0; r < 8; ++r) {
    int k = r + 8 * hi;
    cam[((size_t)n * KCLS + k) * HWP + pix] = acc0[r];
    int k2 = 16 + r + 8 * hi;
    if (k2 < KCLS)
      cam[((size_t)n * KCLS + k2) * HWP + pix] = acc1[r];
  }
}

// ---------------------------------------------------------------------------
// Kernel 2: per-(n,k) mean (score) and max(relu(cam)) over pixels
// grid (20, 32), block 256
// ---------------------------------------------------------------------------
__global__ __launch_bounds__(256) void reduce_kernel(
    const float* __restrict__ cam, float* __restrict__ score,
    float* __restrict__ cammax)
{
  const int k = blockIdx.x, n = blockIdx.y, t = threadIdx.x;
  const float* c = cam + ((size_t)n * KCLS + k) * HWP;
  float s = 0.f, mx = -1e30f;
  for (int p = t; p < HWP; p += 256) {
    float v = c[p];
    s += v;
    mx = fmaxf(mx, v);
  }
  __shared__ float ss[256], sm[256];
  ss[t] = s; sm[t] = mx;
  __syncthreads();
  for (int o = 128; o > 0; o >>= 1) {
    if (t < o) { ss[t] += ss[t + o]; sm[t] = fmaxf(sm[t], sm[t + o]); }
    __syncthreads();
  }
  if (t == 0) {
    score[n * KCLS + k]  = ss[0] * (1.f / HWP);
    cammax[n * KCLS + k] = fmaxf(sm[0], 0.f);  // max of relu == relu of max
  }
}

// ---------------------------------------------------------------------------
// Kernel 3: ncam (bkg + normalized relu cam) * valid_mask, pixel mask
// grid 32, block 1024 (one thread per pixel)
// ---------------------------------------------------------------------------
__global__ __launch_bounds__(1024) void ncam_kernel(
    const float* __restrict__ cam, const float* __restrict__ cammax,
    const float* __restrict__ vm, float* __restrict__ ncam_out,
    float* __restrict__ maskp)
{
  const int n = blockIdx.x;
  const int p = threadIdx.x;
  __shared__ float cmax[KCLS];
  if (p < KCLS) cmax[p] = cammax[n * KCLS + p];
  __syncthreads();

  float nk[KCLS];
  float mx = 0.f;
#pragma unroll
  for (int k = 0; k < KCLS; ++k) {
    float v = fmaxf(cam[((size_t)n * KCLS + k) * HWP + p], 0.f);
    float q = v / (cmax[k] + 1e-5f);
    nk[k] = q;
    mx = fmaxf(mx, q);
  }
  const float bkg = 1.f - mx;
  const size_t base = (size_t)n * 21 * HWP + p;
  float v0 = bkg * vm[base];
  ncam_out[base] = v0;
  float sum = v0;
#pragma unroll
  for (int k = 0; k < KCLS; ++k) {
    float v = nk[k] * vm[base + (size_t)(k + 1) * HWP];
    ncam_out[base + (size_t)(k + 1) * HWP] = v;
    sum += v;
  }
  maskp[n * HWP + p] = (sum > 0.5f) ? 1.f : 0.f;
}

// ---------------------------------------------------------------------------
// Kernel 4: per-image soft clustering (10 iters) + prototype activation
// grid 32, block 256 (8 waves). Centers + assignment matrix live in LDS.
// ---------------------------------------------------------------------------
__device__ __forceinline__ float wred(float v) {
#pragma unroll
  for (int o = 16; o > 0; o >>= 1) v += __shfl_xor(v, o, 32);
  return v;
}

__global__ __launch_bounds__(256) void cluster_kernel(
    const float* __restrict__ hie, const float* __restrict__ maskp,
    const int* __restrict__ seeds, const float* __restrict__ vm,
    float* __restrict__ iscam)
{
  const int n    = blockIdx.x;
  const int tid  = threadIdx.x;
  const int lane = tid & 31;
  const int wv   = tid >> 5;
  const float* feat = hie + (size_t)n * CFC * HWP;

  __shared__ float cf[NSEED * CFC];    // feature centers      (10 KB)
  __shared__ float cfn[NSEED * CFC];   // next centers         (10 KB)
  __shared__ float aArr[NSEED * HWP];  // soft assignments     (20 KB)
  __shared__ float sqf[HWP];           // |feat_p|^2           (4 KB)
  __shared__ float cc[2 * NSEED], ccn[2 * NSEED];
  __shared__ float sqcf[NSEED], asum[NSEED];

  // init: seed feature centers + seed coords
  for (int idx = tid; idx < NSEED * CFC; idx += 256) {
    int s = idx >> 9;           // / 512
    int c = idx & (CFC - 1);
    int sh = seeds[(n * NSEED + s) * 2 + 0];
    int sw = seeds[(n * NSEED + s) * 2 + 1];
    cf[s * CFC + c] = feat[(size_t)c * HWP + sh * WDIM + sw];
  }
  if (tid < NSEED) {
    cc[tid * 2 + 0] = (float)seeds[(n * NSEED + tid) * 2 + 0];
    cc[tid * 2 + 1] = (float)seeds[(n * NSEED + tid) * 2 + 1];
  }
  // per-pixel squared norms
  for (int i = 0; i < 4; ++i) {
    int p = tid + i * 256;
    float acc = 0.f;
    for (int c = 0; c < CFC; ++c) {
      float f = feat[(size_t)c * HWP + p];
      acc += f * f;
    }
    sqf[p] = acc;
  }
  __syncthreads();

  for (int it = 0; it < NITER; ++it) {
    // |cf_s|^2 via wave-cooperative reduce; zero accumulators
    if (wv < NSEED) {
      float acc = 0.f;
      for (int c = lane; c < CFC; c += 32) { float v = cf[wv * CFC + c]; acc += v * v; }
      acc = wred(acc);
      if (lane == 0) sqcf[wv] = acc;
    }
    if (tid < NSEED) asum[tid] = 0.f;
    if (tid < 2 * NSEED) ccn[tid] = 0.f;
    __syncthreads();

    // phase 1: distances -> soft assignments (masked), partial sums
    float psum[NSEED] = {0.f, 0.f, 0.f, 0.f, 0.f};
    for (int i = 0; i < 4; ++i) {
      int p = tid + i * 256;
      float fh = (float)(p >> 5);
      float fw = (float)(p & 31);
      float mval = maskp[n * HWP + p];
      float dot[NSEED] = {0.f, 0.f, 0.f, 0.f, 0.f};
      for (int c = 0; c < CFC; ++c) {
        float f = feat[(size_t)c * HWP + p];
#pragma unroll
        for (int s = 0; s < NSEED; ++s) dot[s] += f * cf[s * CFC + c];
      }
#pragma unroll
      for (int s = 0; s < NSEED; ++s) {
        float dh = fh - cc[s * 2 + 0];
        float dw = fw - cc[s * 2 + 1];
        float sd = dh * dh + dw * dw + 2e-8f;
        float fd = sqf[p] + sqcf[s] - 2.f * dot[s] + 5.12e-6f;   // eps_f = C*1e-8
        float tot = sqrtf(fmaxf(fd + sd * (1.f / 3.f), 1e-12f));
        float av = __expf(-tot) * mval;
        aArr[s * HWP + p] = av;
        psum[s] += av;
      }
    }
#pragma unroll
    for (int s = 0; s < NSEED; ++s) atomicAdd(&asum[s], psum[s]);
    __syncthreads();

    // normalize assignments; accumulate coordinate centers
    float inv[NSEED];
#pragma unroll
    for (int s = 0; s < NSEED; ++s) inv[s] = 1.f / asum[s];
    float hsum[NSEED] = {0.f, 0.f, 0.f, 0.f, 0.f};
    float wsum[NSEED] = {0.f, 0.f, 0.f, 0.f, 0.f};
    for (int i = 0; i < 4; ++i) {
      int p = tid + i * 256;
      float fh = (float)(p >> 5);
      float fw = (float)(p & 31);
#pragma unroll
      for (int s = 0; s < NSEED; ++s) {
        float an = aArr[s * HWP + p] * inv[s];
        aArr[s * HWP + p] = an;
        hsum[s] += fh * an;
        wsum[s] += fw * an;
      }
    }
#pragma unroll
    for (int s = 0; s < NSEED; ++s) {
      atomicAdd(&ccn[s * 2 + 0], hsum[s]);
      atomicAdd(&ccn[s * 2 + 1], wsum[s]);
    }
    __syncthreads();

    // phase 2: new feature centers cfn[s][c] = sum_p feat[c][p]*a[s][p]
    for (int c = wv; c < CFC; c += 8) {
      const float* fr = feat + (size_t)c * HWP;
      float acc[NSEED] = {0.f, 0.f, 0.f, 0.f, 0.f};
      for (int p = lane; p < HWP; p += 32) {
        float f = fr[p];
#pragma unroll
        for (int s = 0; s < NSEED; ++s) acc[s] += f * aArr[s * HWP + p];
      }
#pragma unroll
      for (int s = 0; s < NSEED; ++s) acc[s] = wred(acc[s]);
      if (lane == 0) {
#pragma unroll
        for (int s = 0; s < NSEED; ++s) cfn[s * CFC + c] = acc[s];
      }
    }
    __syncthreads();

    for (int idx = tid; idx < NSEED * CFC; idx += 256) cf[idx] = cfn[idx];
    if (tid < 2 * NSEED) cc[tid] = ccn[tid];
    __syncthreads();
  }

  // prototype inverse norms
  if (wv < NSEED) {
    float acc = 0.f;
    for (int c = lane; c < CFC; c += 32) { float v = cf[wv * CFC + c]; acc += v * v; }
    acc = wred(acc);
    if (lane == 0) sqcf[wv] = 1.f / (sqrtf(acc) + 1e-8f);
  }
  __syncthreads();

  // act = relu(mean_s <pn_s, fn_p>) ; is_cam = act * valid_mask
  for (int i = 0; i < 4; ++i) {
    int p = tid + i * 256;
    float dot[NSEED] = {0.f, 0.f, 0.f, 0.f, 0.f};
    for (int c = 0; c < CFC; ++c) {
      float f = feat[(size_t)c * HWP + p];
#pragma unroll
      for (int s = 0; s < NSEED; ++s) dot[s] += f * cf[s * CFC + c];
    }
    float acc = 0.f;
#pragma unroll
    for (int s = 0; s < NSEED; ++s) acc += dot[s] * sqcf[s];
    float fni = 1.f / (sqrtf(sqf[p]) + 1e-8f);
    float act = fmaxf(acc * (1.f / NSEED) * fni, 0.f);
    size_t base = (size_t)n * 21 * HWP + p;
#pragma unroll
    for (int ch = 0; ch < 21; ++ch)
      iscam[base + (size_t)ch * HWP] = act * vm[base + (size_t)ch * HWP];
  }
}

// ---------------------------------------------------------------------------
extern "C" void kernel_launch(void* const* d_in, const int* in_sizes, int n_in,
                              void* d_out, int out_size, void* d_ws, size_t ws_size,
                              hipStream_t stream) {
  const float* x4    = (const float*)d_in[0];
  const float* hie   = (const float*)d_in[1];
  const float* vm    = (const float*)d_in[2];
  const float* clsw  = (const float*)d_in[3];
  const int*   seeds = (const int*)d_in[4];

  float* out       = (float*)d_out;
  float* score_out = out;                       // 32*20           = 640
  float* ncam_out  = out + 640;                 // 32*21*1024      = 688128
  float* iscam_out = out + 640 + 688128;        // 32*21*1024

  float* ws     = (float*)d_ws;
  float* cam    = ws;                           // 32*20*1024 = 655360 floats
  float* cammax = ws + 655360;                  // 640 floats
  float* maskp  = ws + 656000;                  // 32*1024 floats

  cam_gemm_kernel<<<dim3(8, 32), 256, 0, stream>>>(x4, clsw, cam);
  reduce_kernel<<<dim3(KCLS, NB), 256, 0, stream>>>(cam, score_out, cammax);
  ncam_kernel<<<dim3(NB), 1024, 0, stream>>>(cam, cammax, vm, ncam_out, maskp);
  cluster_kernel<<<dim3(NB), 256, 0, stream>>>(hie, maskp, seeds, vm, iscam_out);
}
